// GptOssTransformerDecoder_71459665871176
// MI455X (gfx1250) — matrix operand order repaired and checked by reading, not compiled, not run on hardware
//
#include <hip/hip_runtime.h>
#include <math.h>

// ---------------- problem constants ----------------
constexpr int B = 2, S = 1024, H = 1024, I = 1024, E = 32;
constexpr int T = B * S;           // 2048 tokens
constexpr int N2 = 2 * I;          // 2048 gate_up cols
constexpr int TOPK = 4;
constexpr float ALPHA = 1.702f;
constexpr float LIMIT = 7.0f;
constexpr int TILE_M = 64;         // token tile per block (4 WMMA m-subtiles)
constexpr int MS = TILE_M / 16;    // 4 m-subtiles -> 4x B-fragment reuse
constexpr int NTILES = T / TILE_M; // 32 (covers worst-case all-tokens->one-expert)

// ---------------- CDNA5 WMMA types ----------------
typedef __attribute__((ext_vector_type(16))) __bf16 v16bf;
typedef __attribute__((ext_vector_type(8)))  __bf16 v8bf;
typedef __attribute__((ext_vector_type(8)))  float  v8f;

// float -> bf16 (round-to-nearest-even), raw-bit helpers
__device__ __forceinline__ unsigned short f2bfu(float f) {
    union { float f; unsigned u; } a; a.f = f;
    unsigned r = a.u + 0x7FFFu + ((a.u >> 16) & 1u);
    return (unsigned short)(r >> 16);
}
__device__ __forceinline__ __bf16 f2bf(float f) {
    union { unsigned short s; __bf16 b; } o; o.s = f2bfu(f);
    return o.b;
}

// Build 16x32 bf16 A-fragment from an LDS-resident row-major bf16 tile.
// ISA layout: lane L -> row M = L&15, half h = L>>4; element j ->
// K = k0 + (j>>3)*16 + h*8 + (j&7)  => two contiguous 16B LDS loads.
__device__ __forceinline__ v16bf load_a_frag(const unsigned short* lds,
                                             int row, int k0, int h, int ld) {
    const v8bf lo = *(const v8bf*)(lds + row * ld + k0 + h * 8);
    const v8bf hi = *(const v8bf*)(lds + row * ld + k0 + 16 + h * 8);
    return __builtin_shufflevector(lo, hi, 0, 1, 2, 3, 4, 5, 6, 7,
                                   8, 9, 10, 11, 12, 13, 14, 15);
}

// ---------------- kernel 1: zero output + expert counters ----------------
__global__ void zero_kernel(float* __restrict__ out, int* __restrict__ cnt) {
    size_t idx = (size_t)blockIdx.x * blockDim.x + threadIdx.x;
    if (idx < (size_t)T * H) out[idx] = 0.0f;
    if (idx < (size_t)E)     cnt[idx] = 0;
}

// ---------------- kernel 2: router (one wave32 per token) ----------------
__global__ void router_kernel(const float* __restrict__ x,
                              const float* __restrict__ rw,
                              const float* __restrict__ rb,
                              float* __restrict__ out_scores,
                              int* __restrict__ cnt,
                              int* __restrict__ tlist,
                              float* __restrict__ slist) {
    const int t = blockIdx.x;        // token
    const int lane = threadIdx.x;    // expert (wave32, E==32)

    const float4* xr = (const float4*)(x + (size_t)t * H);
    const float4* wr = (const float4*)(rw + (size_t)lane * H);
    float acc = 0.0f;
    for (int q = 0; q < H / 4; ++q) {
        float4 a = xr[q], b = wr[q];
        acc = fmaf(a.x, b.x, acc); acc = fmaf(a.y, b.y, acc);
        acc = fmaf(a.z, b.z, acc); acc = fmaf(a.w, b.w, acc);
    }
    acc += rb[lane];

    // top-4 via 4 rounds of wave32 shuffle-argmax (tie-break: lowest index)
    float v = acc;
    float topv[TOPK]; int topi[TOPK];
#pragma unroll
    for (int k = 0; k < TOPK; ++k) {
        float m = v; int mi = lane;
#pragma unroll
        for (int off = 16; off > 0; off >>= 1) {
            float ov = __shfl_xor(m, off, 32);
            int   oi = __shfl_xor(mi, off, 32);
            if (ov > m || (ov == m && oi < mi)) { m = ov; mi = oi; }
        }
        topv[k] = m; topi[k] = mi;
        if (lane == mi) v = -INFINITY;
    }

    float mx = topv[0], sum = 0.0f, p[TOPK];
#pragma unroll
    for (int k = 0; k < TOPK; ++k) { p[k] = __expf(topv[k] - mx); sum += p[k]; }
    float inv = 1.0f / sum;

    float s = 0.0f;
#pragma unroll
    for (int k = 0; k < TOPK; ++k)
        if (topi[k] == lane) s = p[k] * inv;

    out_scores[(size_t)t * E + lane] = s;

    if (s > 0.0f) {
        int pos = atomicAdd(&cnt[lane], 1);
        tlist[(size_t)lane * T + pos] = t;
        slist[(size_t)lane * T + pos] = s;
    }
}

// ---------------- kernel 3: MoE expert GEMMs via bf16 WMMA ----------------
// grid = (NTILES, E), block = 256 (8 wave32).
// LDS: X tile 64x1024 bf16 (128KB) + gated tile (128KB) = 256KB of 320KB/WGP.
__global__ void __launch_bounds__(256)
moe_kernel(const float* __restrict__ x,
           const float* __restrict__ gup,   // [E][H][2I]
           const float* __restrict__ gub,   // [E][2I]
           const float* __restrict__ dpr,   // [E][I][H]
           const float* __restrict__ dbi,   // [E][H]
           const int*   __restrict__ cnt,
           const int*   __restrict__ tlist,
           const float* __restrict__ slist,
           float* __restrict__ out) {
    const int e    = blockIdx.y;
    const int base = blockIdx.x * TILE_M;
    const int ce   = cnt[e];
    if (base >= ce) return;

    __shared__ unsigned short Xs[TILE_M * H];   // gathered tokens, bf16 (128KB)
    __shared__ unsigned short Gs[TILE_M * I];   // gated intermediates (128KB)
    __shared__ int   tokS[TILE_M];
    __shared__ float scS[TILE_M];

    const int tid = threadIdx.x;
    if (tid < TILE_M) {
        int idx = base + tid;
        bool valid = idx < ce;
        tokS[tid] = tlist[(size_t)e * T + (valid ? idx : base)];
        scS[tid]  = valid ? slist[(size_t)e * T + idx] : 0.0f;
    }
    __syncthreads();

    // ---- stage gathered X rows as bf16 (coalesced float4 reads) ----
    for (int u = tid; u < TILE_M * (H / 4); u += 256) {
        int row = u >> 8;            // H/4 = 256 float4 per row
        int q   = u & 255;
        float4 v = *(const float4*)(x + (size_t)tokS[row] * H + q * 4);
        Xs[row * H + q * 4 + 0] = f2bfu(v.x);
        Xs[row * H + q * 4 + 1] = f2bfu(v.y);
        Xs[row * H + q * 4 + 2] = f2bfu(v.z);
        Xs[row * H + q * 4 + 3] = f2bfu(v.w);
    }
    __syncthreads();

    const int wave = tid >> 5;
    const int lane = tid & 31;
    const int c = lane & 15;         // B/D column index, A row index
    const int h = lane >> 4;         // half-wave
    const float* gup_e = gup + (size_t)e * H * N2;
    const float* dpr_e = dpr + (size_t)e * I * H;

    // ---- phase 1: gate_up GEMM + activation -> Gs ----
    // wave owns intermediates [wave*128, wave*128+128); each B fragment
    // (gate & up columns) is reused across MS=4 A fragments (4 WMMAs each).
    for (int it = 0; it < 8; ++it) {
        const int i0   = wave * 128 + it * 16;
        const int colg = 2 * (i0 + c);        // even: gate; +1: up
        v8f accg[MS] = {}, accu[MS] = {};
        const float* bp = gup_e + colg;
        for (int k0 = 0; k0 < H; k0 += 32) {
            __builtin_prefetch((const void*)(bp + (size_t)(k0 + 32 + h * 16) * N2), 0, 0);
            v16bf bg, bu;
#pragma unroll
            for (int j = 0; j < 16; ++j) {
                int K = k0 + (j >> 3) * 16 + h * 8 + (j & 7);      // B layout: lane col N, K = k0+(j>>3)*16+h*8+(j&7)
                float2 p = *(const float2*)(bp + (size_t)K * N2);
                bg[j] = f2bf(p.x);
                bu[j] = f2bf(p.y);
            }
#pragma unroll
            for (int ms = 0; ms < MS; ++ms) {
                v16bf a = load_a_frag(Xs, ms * 16 + c, k0, h, H);
                accg[ms] = __builtin_amdgcn_wmma_f32_16x16x32_bf16(
                    false, a, false, bg, (short)0, accg[ms], false, false);
                accu[ms] = __builtin_amdgcn_wmma_f32_16x16x32_bf16(
                    false, a, false, bu, (short)0, accu[ms], false, false);
            }
        }
        const float gb = gub[(size_t)e * N2 + colg];
        const float ub = gub[(size_t)e * N2 + colg + 1];
#pragma unroll
        for (int ms = 0; ms < MS; ++ms) {
#pragma unroll
            for (int v = 0; v < 8; ++v) {
                float g = accg[ms][v] + gb;
                float u = accu[ms][v] + ub;
                g = fminf(g, LIMIT);
                u = fminf(fmaxf(u, -LIMIT), LIMIT);
                float glu = g / (1.0f + __expf(-ALPHA * g));
                float gated = (u + 1.0f) * glu;
                int m = ms * 16 + v + 8 * h;  // D layout row
                Gs[m * I + i0 + c] = f2bfu(gated);
            }
        }
    }
    __syncthreads();

    // ---- phase 2: down GEMM, scale by score, atomic combine ----
    for (int nt = 0; nt < 8; ++nt) {
        const int col = wave * 128 + nt * 16 + c;
        v8f acc[MS] = {};
        const float* bp = dpr_e + col;
        for (int k0 = 0; k0 < I; k0 += 32) {
            __builtin_prefetch((const void*)(bp + (size_t)(k0 + 32 + h * 16) * H), 0, 0);
            v16bf b;
#pragma unroll
            for (int j = 0; j < 16; ++j) {
                int K = k0 + (j >> 3) * 16 + h * 8 + (j & 7);
                b[j] = f2bf(bp[(size_t)K * H]);
            }
#pragma unroll
            for (int ms = 0; ms < MS; ++ms) {
                v16bf a = load_a_frag(Gs, ms * 16 + c, k0, h, I);
                acc[ms] = __builtin_amdgcn_wmma_f32_16x16x32_bf16(
                    false, a, false, b, (short)0, acc[ms], false, false);
            }
        }
        const float db = dbi[(size_t)e * H + col];
#pragma unroll
        for (int ms = 0; ms < MS; ++ms) {
#pragma unroll
            for (int v = 0; v < 8; ++v) {
                int m = ms * 16 + v + 8 * h;
                float val = scS[m] * (acc[ms][v] + db);
                atomicAdd(&out[(size_t)tokS[m] * H + col], val);
            }
        }
    }
}

// ---------------- host launcher ----------------
extern "C" void kernel_launch(void* const* d_in, const int* in_sizes, int n_in,
                              void* d_out, int out_size, void* d_ws, size_t ws_size,
                              hipStream_t stream) {
    const float* x   = (const float*)d_in[0];  // hidden_states  [B,S,H]
    const float* rw  = (const float*)d_in[1];  // router_weight  [E,H]
    const float* rb  = (const float*)d_in[2];  // router_bias    [E]
    const float* gup = (const float*)d_in[3];  // gate_up_proj   [E,H,2I]
    const float* gub = (const float*)d_in[4];  // gate_up_bias   [E,2I]
    const float* dpr = (const float*)d_in[5];  // down_proj      [E,I,H]
    const float* dbi = (const float*)d_in[6];  // down_bias      [E,H]

    float* out        = (float*)d_out;                 // combined [B,S,H]
    float* out_scores = out + (size_t)T * H;           // router_scores [T,E]

    char* w = (char*)d_ws;
    int*   cnt   = (int*)w;                            // [E]
    int*   tlist = (int*)(w + 128);                    // [E][T]
    float* slist = (float*)(w + 128 + sizeof(int) * (size_t)E * T);

    zero_kernel<<<(T * H) / 256, 256, 0, stream>>>(out, cnt);
    router_kernel<<<T, 32, 0, stream>>>(x, rw, rb, out_scores, cnt, tlist, slist);
    moe_kernel<<<dim3(NTILES, E), 256, 0, stream>>>(x, gup, gub, dpr, dbi,
                                                    cnt, tlist, slist, out);
}
